// Attention_37288906064024
// MI455X (gfx1250) — compile-verified, matrix operand order checked by reading, not run on hardware
//
#include <hip/hip_runtime.h>
#include <cstdint>
#include <cstddef>

// ---------- problem constants ----------
#define BATCH    2
#define SEQ      2048
#define N_EMBD   1024
#define N_HEAD   16
#define HEAD_DIM 64
#define BS       (BATCH * SEQ)

// ---------- vector types for WMMA ----------
typedef __attribute__((ext_vector_type(8)))  float  v8f;
typedef __attribute__((ext_vector_type(16))) __bf16 v16bf;
typedef __attribute__((ext_vector_type(8)))  __bf16 v8bf;

// ---------- fp32 <-> bf16 helpers ----------
__device__ __forceinline__ unsigned short f32_to_bf16(float f) {
    union { float f; unsigned u; } v; v.f = f;
    unsigned u = v.u;
    unsigned r = u + 0x7FFFu + ((u >> 16) & 1u);
    return (unsigned short)(r >> 16);
}
__device__ __forceinline__ float bf16_to_f32(unsigned short h) {
    union { unsigned u; float f; } v; v.u = ((unsigned)h) << 16; return v.f;
}
__device__ __forceinline__ unsigned pack_bf16x2(float a, float b) {
    return (unsigned)f32_to_bf16(a) | ((unsigned)f32_to_bf16(b) << 16);
}

// ---------- bulk fp32 -> bf16 converter ----------
__global__ __launch_bounds__(256)
void cvt_f32_to_bf16(const float* __restrict__ in, unsigned short* __restrict__ out, int n4) {
    int i = blockIdx.x * blockDim.x + threadIdx.x;
    if (i < n4) {
        float4 v = ((const float4*)in)[i];
        uint2 o;
        o.x = pack_bf16x2(v.x, v.y);
        o.y = pack_bf16x2(v.z, v.w);
        ((uint2*)out)[i] = o;
    }
}

// ---------- WMMA 16-bit fragment load from LDS (ISA 7.12.2 layout) ----------
__device__ __forceinline__ v16bf load_frag(const unsigned short* s, int row, int koff,
                                           int lane, int pitch) {
    const int lo = lane & 15;
    const int hi = lane >> 4;
    const unsigned short* r = s + (row + lo) * pitch + koff + hi * 8;
    v8bf c0 = *(const v8bf*)(r);
    v8bf c1 = *(const v8bf*)(r + 16);
    return __builtin_shufflevector(c0, c1, 0,1,2,3,4,5,6,7,8,9,10,11,12,13,14,15);
}

// ---------- CDNA5 async copy: global -> LDS, 16B per lane, ASYNCcnt tracked ----------
__device__ __forceinline__ void async_copy_b128(const void* gptr, void* lptr) {
    const unsigned      ldsoff = (unsigned)(size_t)lptr;        // low 32b of flat LDS addr
    const unsigned long long ga = (unsigned long long)(size_t)gptr;
    asm volatile("global_load_async_to_lds_b128 %0, %1, off"
                 :: "v"(ldsoff), "v"(ga) : "memory");
}
__device__ __forceinline__ void async_wait_le4() {
    asm volatile("s_wait_asynccnt 0x4" ::: "memory");
}
__device__ __forceinline__ void async_wait_0() {
    asm volatile("s_wait_asynccnt 0x0" ::: "memory");
}

// ---------- output scatter layouts for the dense GEMM ----------
enum { L_ROWMAJOR = 0, L_QKV = 1, L_VT = 2 };

// C = A (M x K) * B^T (N x K) [+ bias] * scale; bf16 WMMA, f32 accum;
// register double-buffered global->LDS staging.
template<int LAYOUT, bool OUT_BF16>
__global__ __launch_bounds__(256)
void gemm_wmma(const unsigned short* __restrict__ A, const unsigned short* __restrict__ B,
               const float* __restrict__ bias, void* __restrict__ Cv,
               int M, int N, int K, int lda, int ldb, int ldc, float scale)
{
    constexpr int BM = 128, BN = 64, BK = 32, PITCH = 40;
    __shared__ alignas(16) unsigned short As[BM * PITCH];
    __shared__ alignas(16) unsigned short Bs[BN * PITCH];

    const int tid  = threadIdx.x;
    const int lane = tid & 31;
    const int wave = tid >> 5;
    const int blockM = blockIdx.y * BM;
    const int blockN = blockIdx.x * BN;

    const int mW = (wave & 3) * 32;
    const int nW = (wave >> 2) * 32;

    v8f acc[2][2] = {};

    const int fillRow   = tid >> 2;
    const int fillChunk = (tid & 3) * 8;

    uint4 aR[2], bR;
    #pragma unroll
    for (int rr = 0; rr < 2; ++rr)
        aR[rr] = *(const uint4*)(A + (size_t)(blockM + fillRow + rr * 64) * lda + fillChunk);
    bR = *(const uint4*)(B + (size_t)(blockN + fillRow) * ldb + fillChunk);

    for (int k0 = 0; k0 < K; k0 += BK) {
        #pragma unroll
        for (int rr = 0; rr < 2; ++rr)
            *(uint4*)(As + (fillRow + rr * 64) * PITCH + fillChunk) = aR[rr];
        *(uint4*)(Bs + fillRow * PITCH + fillChunk) = bR;
        __syncthreads();

        if (k0 + BK < K) {   // issue next tile early; latency hides behind WMMA
            const int kn = k0 + BK + fillChunk;
            #pragma unroll
            for (int rr = 0; rr < 2; ++rr)
                aR[rr] = *(const uint4*)(A + (size_t)(blockM + fillRow + rr * 64) * lda + kn);
            bR = *(const uint4*)(B + (size_t)(blockN + fillRow) * ldb + kn);
        }

        v16bf aF[2], bF[2];
        #pragma unroll
        for (int i = 0; i < 2; ++i) aF[i] = load_frag(As, mW + i * 16, 0, lane, PITCH);
        #pragma unroll
        for (int j = 0; j < 2; ++j) bF[j] = load_frag(Bs, nW + j * 16, 0, lane, PITCH);

        #pragma unroll
        for (int i = 0; i < 2; ++i)
            #pragma unroll
            for (int j = 0; j < 2; ++j)
                acc[i][j] = __builtin_amdgcn_wmma_f32_16x16x32_bf16(
                    false, aF[i], false, bF[j], (short)0, acc[i][j], false, false);

        __syncthreads();
    }

    const int lo = lane & 15;
    const int hi = lane >> 4;
    #pragma unroll
    for (int i = 0; i < 2; ++i) {
        #pragma unroll
        for (int j = 0; j < 2; ++j) {
            const int n = blockN + nW + j * 16 + lo;
            const float bn = bias ? bias[n] : 0.0f;
            #pragma unroll
            for (int r = 0; r < 8; ++r) {
                const int m = blockM + mW + i * 16 + hi * 8 + r;
                const float v = (acc[i][j][r] + bn) * scale;
                size_t addr;
                if (LAYOUT == L_ROWMAJOR) {
                    addr = (size_t)m * ldc + n;
                } else if (LAYOUT == L_QKV) {          // [B,H,S,Dh]
                    const int b = m >> 11, s_ = m & 2047, h = n >> 6, d = n & 63;
                    addr = ((((size_t)b * N_HEAD + h) * SEQ + s_) << 6) + d;
                } else {                               // L_VT: [B,H,Dh,S]
                    const int b = m >> 11, s_ = m & 2047, h = n >> 6, d = n & 63;
                    addr = ((((size_t)b * N_HEAD + h) * HEAD_DIM + d) << 11) + s_;
                }
                if (OUT_BF16) ((unsigned short*)Cv)[addr] = f32_to_bf16(v);
                else          ((float*)Cv)[addr] = v;
            }
        }
    }
}

// =====================================================================
// Fused attention core (64 q-rows of one (b,h) per WG).
// K/V tiles staged with CDNA5 async-to-LDS copies, double buffered.
// Score stripe (64 x 2048 bf16) stays resident in LDS across
// QK^T -> softmax -> P·V.  Dynamic LDS ~311 KB (<320 KB WGP budget).
// =====================================================================
#define WPITCH 2056               // 2048 + 8 pad
#define KPITCH 72                 // 64 + 8 pad
#define VPITCH 136                // 128 + 8 pad
#define STG_HALF (128 * KPITCH)   // one staging buffer (elements); V tile (64*VPITCH) fits too
#define FUSED_SMEM ((64*KPITCH + 2*STG_HALF + 64*WPITCH) * 2 + (256 + 64 + 64) * 4)

__global__ __launch_bounds__(256)
void attn_fused(const unsigned short* __restrict__ Q,   // [B,H,S,Dh] bf16 (pre-scaled 1/8)
                const unsigned short* __restrict__ Km,  // [B,H,S,Dh] bf16
                const unsigned short* __restrict__ Vt,  // [B,H,Dh,S] bf16
                float* __restrict__ Wout,               // [B,H,S,S] fp32 (d_out)
                unsigned short* __restrict__ Attn)      // [B,S,E] bf16
{
    extern __shared__ char smem[];
    unsigned short* Sq  = (unsigned short*)smem;        // [64][KPITCH]
    unsigned short* Stg = Sq  + 64 * KPITCH;            // [2][STG_HALF]
    unsigned short* Sw  = Stg + 2 * STG_HALF;           // [64][WPITCH]
    float* pred   = (float*)(Sw + 64 * WPITCH);
    float* rowMax = pred + 256;
    float* rowInv = rowMax + 64;

    const int tid  = threadIdx.x;
    const int lane = tid & 31;
    const int wave = tid >> 5;
    const int lo   = lane & 15;
    const int hi   = lane >> 4;

    const int z  = blockIdx.y;                 // b*16 + h
    const int q0 = blockIdx.x * 64;
    const unsigned short* Qz = Q  + (size_t)z * SEQ * HEAD_DIM;
    const unsigned short* Kz = Km + (size_t)z * SEQ * HEAD_DIM;
    const unsigned short* Vz = Vt + (size_t)z * HEAD_DIM * SEQ;

    // K tile (128 x 64) async issue: 4 x b128 per thread
    auto issueK = [&](int kt, int buf) {
        unsigned short* dst = Stg + buf * STG_HALF;
        #pragma unroll
        for (int p = 0; p < 4; ++p) {
            const int c = tid + p * 256;
            const int r = c >> 3, k = (c & 7) * 8;
            async_copy_b128(Kz + (size_t)(kt * 128 + r) * HEAD_DIM + k,
                            dst + r * KPITCH + k);
        }
    };
    // V^T tile (64 x 128) async issue
    auto issueV = [&](int kt, int buf) {
        unsigned short* dst = Stg + buf * STG_HALF;
        #pragma unroll
        for (int p = 0; p < 4; ++p) {
            const int c = tid + p * 256;
            const int r = c >> 4, k = (c & 15) * 8;
            async_copy_b128(Vz + (size_t)r * SEQ + kt * 128 + k,
                            dst + r * VPITCH + k);
        }
    };

    // ---- phase A: stage Q tile (64 x 64) + kick off first K tile ----
    issueK(0, 0);
    #pragma unroll
    for (int p = 0; p < 2; ++p) {
        const int c = tid + p * 256;
        const int r = c >> 3, k = (c & 7) * 8;
        *(uint4*)(Sq + r * KPITCH + k) = *(const uint4*)(Qz + (size_t)(q0 + r) * HEAD_DIM + k);
    }

    // ---- phase B: scores -> LDS stripe ----
    const int smi = wave & 3;
    const int snb = (wave >> 2) * 64;
    for (int kt = 0; kt < 16; ++kt) {
        if (kt < 15) { issueK(kt + 1, (kt + 1) & 1); async_wait_le4(); }
        else         { async_wait_0(); }
        __syncthreads();                        // tile kt visible to all waves
        const unsigned short* Skv = Stg + (kt & 1) * STG_HALF;

        v8f acc[4] = {};
        #pragma unroll
        for (int ks = 0; ks < 64; ks += 32) {
            v16bf aF = load_frag(Sq, smi * 16, ks, lane, KPITCH);
            #pragma unroll
            for (int j = 0; j < 4; ++j) {
                v16bf bF = load_frag(Skv, snb + j * 16, ks, lane, KPITCH);
                acc[j] = __builtin_amdgcn_wmma_f32_16x16x32_bf16(
                    false, aF, false, bF, (short)0, acc[j], false, false);
            }
        }
        #pragma unroll
        for (int j = 0; j < 4; ++j)
            #pragma unroll
            for (int r = 0; r < 8; ++r) {
                const int row = smi * 16 + hi * 8 + r;
                const int col = kt * 128 + snb + j * 16 + lo;
                Sw[row * WPITCH + col] = f32_to_bf16(acc[j][r]);
            }
        __syncthreads();                        // buffer may be overwritten next iter
    }

    // ---- phase C: softmax over the stripe (vectorized 8 bf16 / ds op) ----
    {
        const int r    = tid >> 2;
        const int cbas = (tid & 3) * 512;
        float mx = -3.0e38f;
        for (int c = 0; c < 512; c += 8) {
            uint4 v = *(const uint4*)(Sw + r * WPITCH + cbas + c);
            const unsigned w[4] = { v.x, v.y, v.z, v.w };
            #pragma unroll
            for (int q = 0; q < 4; ++q) {
                mx = fmaxf(mx, bf16_to_f32((unsigned short)(w[q] & 0xffff)));
                mx = fmaxf(mx, bf16_to_f32((unsigned short)(w[q] >> 16)));
            }
        }
        pred[tid] = mx;
        __syncthreads();
        if (tid < 64)
            rowMax[tid] = fmaxf(fmaxf(pred[tid*4], pred[tid*4+1]),
                                fmaxf(pred[tid*4+2], pred[tid*4+3]));
        __syncthreads();
        const float rm = rowMax[r];
        float sum = 0.f;
        for (int c = 0; c < 512; c += 8) {
            unsigned short* p8 = Sw + r * WPITCH + cbas + c;
            uint4 v = *(const uint4*)p8;
            unsigned w[4] = { v.x, v.y, v.z, v.w };
            #pragma unroll
            for (int q = 0; q < 4; ++q) {
                const float e0 = __expf(bf16_to_f32((unsigned short)(w[q] & 0xffff)) - rm);
                const float e1 = __expf(bf16_to_f32((unsigned short)(w[q] >> 16)) - rm);
                sum += e0 + e1;
                w[q] = pack_bf16x2(e0, e1);
            }
            *(uint4*)p8 = make_uint4(w[0], w[1], w[2], w[3]);   // stripe := exp()
        }
        pred[tid] = sum;
        __syncthreads();
        if (tid < 64)
            rowInv[tid] = 1.0f / (pred[tid*4] + pred[tid*4+1] + pred[tid*4+2] + pred[tid*4+3]);
        __syncthreads();
    }
    // normalize stripe + the single fp32 weights write (float2 / packed bf16x2)
    {
        float* Wz = Wout + (size_t)z * SEQ * SEQ + (size_t)q0 * SEQ;
        for (int idx = tid; idx < 64 * 1024; idx += 256) {
            const int r = idx >> 10, c2 = (idx & 1023) * 2;
            unsigned short* p2 = Sw + r * WPITCH + c2;
            const unsigned w = *(const unsigned*)p2;
            const float inv = rowInv[r];
            const float w0 = bf16_to_f32((unsigned short)(w & 0xffff)) * inv;
            const float w1 = bf16_to_f32((unsigned short)(w >> 16)) * inv;
            *(float2*)(Wz + (size_t)r * SEQ + c2) = make_float2(w0, w1);
            *(unsigned*)p2 = pack_bf16x2(w0, w1);
        }
    }
    __syncthreads();

    // ---- phase D: out = P (64 x 2048) * V (2048 x 64) ----
    const int pmi = wave & 3;
    const int pnb = (wave >> 2) * 2;
    v8f oacc[2] = {};
    issueV(0, 0);
    for (int kt = 0; kt < 16; ++kt) {
        if (kt < 15) { issueV(kt + 1, (kt + 1) & 1); async_wait_le4(); }
        else         { async_wait_0(); }
        __syncthreads();
        const unsigned short* Svt = Stg + (kt & 1) * STG_HALF;

        #pragma unroll
        for (int ks = 0; ks < 128; ks += 32) {
            v16bf aF = load_frag(Sw, pmi * 16, kt * 128 + ks, lane, WPITCH);
            #pragma unroll
            for (int j = 0; j < 2; ++j) {
                v16bf bF = load_frag(Svt, (pnb + j) * 16, ks, lane, VPITCH);
                oacc[j] = __builtin_amdgcn_wmma_f32_16x16x32_bf16(
                    false, aF, false, bF, (short)0, oacc[j], false, false);
            }
        }
        __syncthreads();
    }

    {
        const int b = z >> 4, h = z & 15;
        #pragma unroll
        for (int j = 0; j < 2; ++j)
            #pragma unroll
            for (int r = 0; r < 8; ++r) {
                const int row = pmi * 16 + hi * 8 + r;
                const int col = (pnb + j) * 16 + lo;
                const size_t addr = ((size_t)(b * SEQ + q0 + row) * N_EMBD)
                                  + (size_t)h * HEAD_DIM + col;
                Attn[addr] = f32_to_bf16(oacc[j][r]);
            }
    }
}

extern "C" void kernel_launch(void* const* d_in, const int* in_sizes, int n_in,
                              void* d_out, int out_size, void* d_ws, size_t ws_size,
                              hipStream_t stream) {
    const float* x   = (const float*)d_in[0];
    const float* Wq  = (const float*)d_in[1];
    const float* bq  = (const float*)d_in[2];
    const float* Wk  = (const float*)d_in[3];
    const float* bk  = (const float*)d_in[4];
    const float* Wv  = (const float*)d_in[5];
    const float* bv  = (const float*)d_in[6];
    const float* Wfc = (const float*)d_in[7];
    const float* bfc = (const float*)d_in[8];

    float* out = (float*)d_out;                               // [2,2048,1024]
    float* wts = out + (size_t)BS * N_EMBD;                   // [2,16,2048,2048]

    unsigned short* X16   = (unsigned short*)d_ws;
    unsigned short* Wq16  = X16   + (size_t)BS * N_EMBD;
    unsigned short* Wk16  = Wq16  + (size_t)N_EMBD * N_EMBD;
    unsigned short* Wv16  = Wk16  + (size_t)N_EMBD * N_EMBD;
    unsigned short* Wfc16 = Wv16  + (size_t)N_EMBD * N_EMBD;
    unsigned short* Q16   = Wfc16 + (size_t)N_EMBD * N_EMBD;  // [B,H,S,Dh]
    unsigned short* K16   = Q16   + (size_t)BS * N_EMBD;
    unsigned short* Vt16  = K16   + (size_t)BS * N_EMBD;      // [B,H,Dh,S]
    unsigned short* Attn16= Vt16  + (size_t)BS * N_EMBD;      // [B,S,E]

    (void)hipFuncSetAttribute((const void*)attn_fused,
                              hipFuncAttributeMaxDynamicSharedMemorySize, FUSED_SMEM);

    {   // fp32 -> bf16 conversions
        long nx = (long)BS * N_EMBD;    int n4 = (int)(nx / 4);
        cvt_f32_to_bf16<<<(n4 + 255) / 256, 256, 0, stream>>>(x, X16, n4);
        long nw = (long)N_EMBD * N_EMBD; n4 = (int)(nw / 4);
        cvt_f32_to_bf16<<<(n4 + 255) / 256, 256, 0, stream>>>(Wq,  Wq16,  n4);
        cvt_f32_to_bf16<<<(n4 + 255) / 256, 256, 0, stream>>>(Wk,  Wk16,  n4);
        cvt_f32_to_bf16<<<(n4 + 255) / 256, 256, 0, stream>>>(Wv,  Wv16,  n4);
        cvt_f32_to_bf16<<<(n4 + 255) / 256, 256, 0, stream>>>(Wfc, Wfc16, n4);
    }

    const dim3 blk(256);

    gemm_wmma<L_QKV, true><<<dim3(16, 32), blk, 0, stream>>>(
        X16, Wq16, bq, Q16, BS, N_EMBD, N_EMBD, N_EMBD, N_EMBD, 0, 0.125f);
    gemm_wmma<L_QKV, true><<<dim3(16, 32), blk, 0, stream>>>(
        X16, Wk16, bk, K16, BS, N_EMBD, N_EMBD, N_EMBD, N_EMBD, 0, 1.0f);
    gemm_wmma<L_VT,  true><<<dim3(16, 32), blk, 0, stream>>>(
        X16, Wv16, bv, Vt16, BS, N_EMBD, N_EMBD, N_EMBD, N_EMBD, 0, 1.0f);

    attn_fused<<<dim3(32, 32), blk, FUSED_SMEM, stream>>>(
        Q16, K16, Vt16, wts, Attn16);

    gemm_wmma<L_ROWMAJOR, false><<<dim3(16, 32), blk, 0, stream>>>(
        Attn16, Wfc16, bfc, out, BS, N_EMBD, N_EMBD, N_EMBD, N_EMBD, N_EMBD, 1.0f);
}
